// ChunkedSelfAttention_73547019976748
// MI455X (gfx1250) — matrix operand order repaired and checked by reading, not compiled, not run on hardware
//
#include <hip/hip_runtime.h>
#include <hip/hip_bf16.h>

// ---------------------------------------------------------------------------
// ChunkedSelfAttention on MI455X (gfx1250, wave32, WMMA).
//   B=2, S=2048, D=1024, H=16, HD=64.  ~103 GFLOP, ~100 MB HBM -> compute
//   bound; all GEMMs + attention through v_wmma_f32_16x16x32_bf16, flash-style
//   online softmax so the [B,H,S,S] score tensor (1 GB) never touches HBM.
//   V is produced pre-transposed per head so the P*V operand loads are
//   contiguous and the attention kernel needs no barriers / V staging.
//   Row-sums come from one WMMA against an all-ones operand (P * 1 = rowsum),
//   halving the ds_bpermute reduction traffic vs a shuffle butterfly.
// ---------------------------------------------------------------------------

typedef __attribute__((ext_vector_type(16))) __bf16 v16bf;
typedef __attribute__((ext_vector_type(8)))  float  v8f;

union FragBF {               // 16 bf16 = 32 bytes = 8 VGPRs (A or B operand)
    v16bf v;
    uint4 u[2];
};

__device__ __forceinline__ v8f wmma_bf16(v16bf a, v16bf b, v8f c) {
    // (neg_a, A, neg_b, B, c_mod, C, reuse_a, reuse_b)
    return __builtin_amdgcn_wmma_f32_16x16x32_bf16(false, a, false, b,
                                                   (short)0, c, false, false);
}

static constexpr int Bn  = 2;
static constexpr int Sn  = 2048;
static constexpr int Dn  = 1024;
static constexpr int Hn  = 16;
static constexpr int HDn = 64;
static constexpr int Mrows = Bn * Sn;        // 4096 flattened (b,s) rows

// ---------------------------------------------------------------------------
// GEMM:  C[M=4096, N=1024] = (A[M,1024] * W[N,1024]^T + bias[N]) * outScale
//   A_F32:      A is f32 (x input) else bf16 (attention output)
//   STORE_MODE: 0 = f32 row-major [M,N]          (final output)
//               1 = bf16 [B,H,S,HD] head-major   (Q and K)
//               2 = bf16 [B,H,HD,S] transposed   (V, for contiguous P*V frags)
// WG = 256 thr = 8 waves; WG tile 128(M) x 64(N); wave tile 32x32 (2x2 frags);
// K staged in LDS in 32-wide steps with f32->bf16 conversion.
// ---------------------------------------------------------------------------
template <bool A_F32, int STORE_MODE>
__global__ __launch_bounds__(256)
void gemm_wmma_kernel(const void* __restrict__ Aptr,
                      const float* __restrict__ W,
                      const float* __restrict__ bias,
                      void* __restrict__ Cptr,
                      float outScale)
{
    constexpr int K = Dn, N = Dn;
    constexpr int TM = 128, TN = 64, TK = 32;
    constexpr int LDSP = 40;                      // row pitch (80B, 16B-aligned)

    __shared__ __attribute__((aligned(16))) __bf16 As[TM][LDSP];
    __shared__ __attribute__((aligned(16))) __bf16 Bs[TN][LDSP];

    const int tid   = threadIdx.x;
    const int wave  = tid >> 5;
    const int lane  = tid & 31;
    const int lr    = lane & 15;
    const int hsel  = (lane >= 16) ? 1 : 0;       // which 16-lane half
    const int waveM = (wave & 3) * 32;            // 4 waves along M
    const int waveN = (wave >> 2) * 32;           // 2 waves along N
    const int bm    = blockIdx.x * TM;
    const int bn    = blockIdx.y * TN;

    v8f acc[2][2];
#pragma unroll
    for (int i = 0; i < 2; ++i)
#pragma unroll
        for (int j = 0; j < 2; ++j)
#pragma unroll
            for (int e = 0; e < 8; ++e) acc[i][j][e] = 0.0f;

    for (int k0 = 0; k0 < K; k0 += TK) {
        // ---- stage A (128x32) and W (64x32) tiles into LDS as bf16 ----
#pragma unroll
        for (int idx = tid; idx < TM * TK; idx += 256) {
            const int r = idx >> 5, c = idx & 31;
            __bf16 val;
            if (A_F32)
                val = (__bf16)(((const float*)Aptr)[(size_t)(bm + r) * K + k0 + c]);
            else
                val = ((const __bf16*)Aptr)[(size_t)(bm + r) * K + k0 + c];
            As[r][c] = val;
        }
#pragma unroll
        for (int idx = tid; idx < TN * TK; idx += 256) {
            const int r = idx >> 5, c = idx & 31;
            Bs[r][c] = (__bf16)W[(size_t)(bn + r) * K + k0 + c];
        }
        __syncthreads();

        // ---- fragment loads (per CDNA5 16-bit A/B VGPR layouts) ----
        FragBF a[2], b[2];
#pragma unroll
        for (int fm = 0; fm < 2; ++fm) {
            const __bf16* rp = &As[waveM + fm * 16 + lr][0];
            const int c0 = hsel ? 8 : 0;          // lanes 16-31: K=8..15,24..31
            a[fm].u[0] = *(const uint4*)(rp + c0);
            a[fm].u[1] = *(const uint4*)(rp + c0 + 16);
        }
#pragma unroll
        for (int fn = 0; fn < 2; ++fn) {
            const __bf16* rp = &Bs[waveN + fn * 16 + lr][0];
            const int kc = hsel ? 8 : 0;          // lanes 16-31: K=8..15,24..31
            b[fn].u[0] = *(const uint4*)(rp + kc);
            b[fn].u[1] = *(const uint4*)(rp + kc + 16);
        }
#pragma unroll
        for (int fm = 0; fm < 2; ++fm)
#pragma unroll
            for (int fn = 0; fn < 2; ++fn)
                acc[fm][fn] = wmma_bf16(a[fm].v, b[fn].v, acc[fm][fn]);

        __syncthreads();
    }

    // ---- epilogue: bias + optional scale + store ----
#pragma unroll
    for (int fm = 0; fm < 2; ++fm) {
#pragma unroll
        for (int fn = 0; fn < 2; ++fn) {
            const int ncol = bn + waveN + fn * 16 + lr;
            const float bv = bias[ncol];
#pragma unroll
            for (int e = 0; e < 8; ++e) {
                const int mrow = bm + waveM + fm * 16 + e + hsel * 8;
                const float val = (acc[fm][fn][e] + bv) * outScale;
                const int h  = ncol >> 6, hd = ncol & 63;
                const int bb = mrow >> 11, s = mrow & 2047;
                if (STORE_MODE == 1) {
                    ((__bf16*)Cptr)[(((size_t)(bb * Hn + h)) * Sn + s) * HDn + hd] =
                        (__bf16)val;
                } else if (STORE_MODE == 2) {
                    ((__bf16*)Cptr)[(((size_t)(bb * Hn + h)) * HDn + hd) * Sn + s] =
                        (__bf16)val;
                } else {
                    ((float*)Cptr)[(size_t)mrow * N + ncol] = val;
                }
            }
        }
    }
}

// ---------------------------------------------------------------------------
// Flash attention per (query-block, head, batch).  Barrier-free.
// WG = 128 thr = 4 waves; each wave owns 16 query rows; 64 q-rows per WG.
// Key loop in blocks of 32: S = Q*K^T (WMMA, scale pre-folded into Q),
// online softmax (row max via intra-half shfl_xor; row sum via a ones-WMMA),
// P routed C-layout -> A-layout through per-wave LDS, O += P*V against the
// pre-transposed V (contiguous global B-frags, no staging).
// ---------------------------------------------------------------------------
__global__ __launch_bounds__(128)
void flash_attn_kernel(const __bf16* __restrict__ Q,
                       const __bf16* __restrict__ Kt,
                       const __bf16* __restrict__ Vt,   // [B,H,HD,S]
                       __bf16* __restrict__ Out)        // [B,S,D] bf16
{
    constexpr int LDSP = 40;
    constexpr int NKB  = Sn / 32;

    __shared__ __attribute__((aligned(16))) __bf16 Pb[4][16][LDSP]; // per-wave P

    const int qb   = blockIdx.x;                  // 0..31 (64 queries each)
    const int h    = blockIdx.y;
    const int bb   = blockIdx.z;
    const int tid  = threadIdx.x;
    const int wave = tid >> 5;
    const int lane = tid & 31;
    const int lr   = lane & 15;
    const int hsel = (lane >= 16) ? 1 : 0;

    const size_t headBase = ((size_t)(bb * Hn + h)) * Sn * HDn;  // same for Vt
    const int qrow0 = qb * 64 + wave * 16;

    // Q fragments (16 rows x 64 hd) pinned in registers: 2 A-frags of 16x32
    FragBF qa[2];
    {
        const __bf16* qp = Q + headBase + (size_t)(qrow0 + lr) * HDn;
        const int c0 = hsel ? 8 : 0;
#pragma unroll
        for (int ks = 0; ks < 2; ++ks) {
            qa[ks].u[0] = *(const uint4*)(qp + ks * 32 + c0);
            qa[ks].u[1] = *(const uint4*)(qp + ks * 32 + c0 + 16);
        }
    }

    // all-ones B operand: P(16x32) * ones(32x16) -> row sums in every column
    FragBF ones;
#pragma unroll
    for (int i = 0; i < 16; ++i) ones.v[i] = (__bf16)1.0f;

    v8f   o[4], osum;
    float rmax[8];
#pragma unroll
    for (int c = 0; c < 4; ++c)
#pragma unroll
        for (int e = 0; e < 8; ++e) o[c][e] = 0.0f;
#pragma unroll
    for (int e = 0; e < 8; ++e) { osum[e] = 0.0f; rmax[e] = -INFINITY; }

    for (int kb = 0; kb < NKB; ++kb) {
        const size_t kBase = headBase + (size_t)kb * 32 * HDn;   // K rows
        // ---- issue ALL operand loads up front so they overlap compute ----
        FragBF bk[2][2];                          // [ks chunk][key half]
#pragma unroll
        for (int half = 0; half < 2; ++half) {
            const __bf16* kp =
                Kt + kBase + (size_t)(half * 16 + lr) * HDn + hsel * 8;
#pragma unroll
            for (int ks = 0; ks < 2; ++ks) {
                bk[ks][half].u[0] = *(const uint4*)(kp + ks * 32);
                bk[ks][half].u[1] = *(const uint4*)(kp + ks * 32 + 16);
            }
        }
        FragBF bv[4];                             // V^T rows: contiguous keys
#pragma unroll
        for (int c = 0; c < 4; ++c) {
            const __bf16* vp =
                Vt + headBase + (size_t)(c * 16 + lr) * Sn + kb * 32 + hsel * 8;
            bv[c].u[0] = *(const uint4*)(vp);
            bv[c].u[1] = *(const uint4*)(vp + 16);
        }
        if (kb + 1 < NKB) {                       // gfx1250 global_prefetch_b8
            __builtin_prefetch(Kt + kBase + 32 * HDn + lane * 64, 0, 1);
        }

        // ---- scores: S(16x32) = Qs * K^T, two 16x16 C-frags ----
        v8f s0, s1;
#pragma unroll
        for (int e = 0; e < 8; ++e) { s0[e] = 0.0f; s1[e] = 0.0f; }
#pragma unroll
        for (int ks = 0; ks < 2; ++ks) {
            s0 = wmma_bf16(qa[ks].v, bk[ks][0].v, s0);
            s1 = wmma_bf16(qa[ks].v, bk[ks][1].v, s1);
        }

        // ---- online softmax; row m = e + hsel*8 lives on one 16-lane half --
#pragma unroll
        for (int e = 0; e < 8; ++e) {
            const float x0 = s0[e];
            const float x1 = s1[e];
            float mloc = fmaxf(x0, x1);
#pragma unroll
            for (int msk = 1; msk < 16; msk <<= 1)
                mloc = fmaxf(mloc, __shfl_xor(mloc, msk, 32));
            const float mnew  = fmaxf(rmax[e], mloc);
            const float alpha = __expf(rmax[e] - mnew);   // 0 on first block
            rmax[e] = mnew;
#pragma unroll
            for (int c = 0; c < 4; ++c) o[c][e] *= alpha;
            osum[e] *= alpha;
            const int row = e + hsel * 8;
            Pb[wave][row][lr]      = (__bf16)__expf(x0 - mnew);
            Pb[wave][row][lr + 16] = (__bf16)__expf(x1 - mnew);
        }

        // ---- reload P as an A-frag (16x32 over keys); same-wave LDS only ---
        FragBF pf;
        {
            const __bf16* pp = &Pb[wave][lr][0];
            const int c0 = hsel ? 8 : 0;
            pf.u[0] = *(const uint4*)(pp + c0);
            pf.u[1] = *(const uint4*)(pp + c0 + 16);
        }

        // ---- O(16x64) += P*V (4 WMMAs) ; row-sum += P*ones (1 WMMA) ----
#pragma unroll
        for (int c = 0; c < 4; ++c)
            o[c] = wmma_bf16(pf.v, bv[c].v, o[c]);
        osum = wmma_bf16(pf.v, ones.v, osum);
    }

    // ---- epilogue: normalize, store bf16 [B,S,D] for output projection ----
    const int grow = bb * Sn + qrow0;
#pragma unroll
    for (int e = 0; e < 8; ++e) {
        const float inv = 1.0f / osum[e];
        const int m = grow + e + hsel * 8;
#pragma unroll
        for (int c = 0; c < 4; ++c) {
            const int col = h * HDn + c * 16 + lr;
            Out[(size_t)m * Dn + col] = (__bf16)(o[c][e] * inv);
        }
    }
}

// ---------------------------------------------------------------------------
extern "C" void kernel_launch(void* const* d_in, const int* in_sizes, int n_in,
                              void* d_out, int out_size, void* d_ws, size_t ws_size,
                              hipStream_t stream)
{
    const float* x  = (const float*)d_in[0];
    const float* Wq = (const float*)d_in[1];
    const float* bq = (const float*)d_in[2];
    const float* Wk = (const float*)d_in[3];
    const float* bk = (const float*)d_in[4];
    const float* Wv = (const float*)d_in[5];
    const float* bv = (const float*)d_in[6];
    const float* Wo = (const float*)d_in[7];
    const float* bo = (const float*)d_in[8];

    // Workspace: Q, K (bf16 [B,H,S,HD]), V (bf16 [B,H,HD,S]), attn [B,S,D]
    const size_t ELEMS = (size_t)Mrows * Dn;      // 4M elements, 8 MB each
    __bf16* Qws = (__bf16*)d_ws;
    __bf16* Kws = Qws + ELEMS;
    __bf16* Vws = Kws + ELEMS;
    __bf16* Aws = Vws + ELEMS;

    const float qScale = 0.125f;                  // 1/sqrt(HD) folded into Q
    const dim3 gGemm(Mrows / 128, Dn / 64);       // 32 x 16
    gemm_wmma_kernel<true, 1><<<gGemm, 256, 0, stream>>>(x, Wq, bq, Qws, qScale);
    gemm_wmma_kernel<true, 1><<<gGemm, 256, 0, stream>>>(x, Wk, bk, Kws, 1.0f);
    gemm_wmma_kernel<true, 2><<<gGemm, 256, 0, stream>>>(x, Wv, bv, Vws, 1.0f);

    flash_attn_kernel<<<dim3(Sn / 64, Hn, Bn), 128, 0, stream>>>(Qws, Kws, Vws, Aws);

    gemm_wmma_kernel<false, 0><<<gGemm, 256, 0, stream>>>(Aws, Wo, bo,
                                                          (float*)d_out, 1.0f);
}
